// EncoderLayer_58497454571711
// MI455X (gfx1250) — compile-verified, matrix operand order checked by reading, not run on hardware
//
#include <hip/hip_runtime.h>

typedef __attribute__((ext_vector_type(16))) __bf16 v16bf;
typedef __attribute__((ext_vector_type(8)))  float  v8f;

#define D_MODEL  1024
#define N_HEADS  16
#define HEAD_DIM 64
#define D_FF     4096
#define SEQ      2048
#define BATCH    2
#define NTOK     (BATCH * SEQ)   // 4096 rows

// ---------------------------------------------------------------------------
// Fragment helpers.
// A-style 16x32 bf16 fragment: lane L holds row M = L&15; K base = (L>>4)*8;
// vector elems e0..e7 = K kb..kb+7, e8..e15 = K kb+16..kb+23  -> two 16B loads.
// ---------------------------------------------------------------------------
__device__ __forceinline__ v16bf ldfrag(const __bf16* p0, const __bf16* p1) {
    union { struct { uint4 lo, hi; } u; v16bf v; } f;
    f.u.lo = *(const uint4*)p0;
    f.u.hi = *(const uint4*)p1;
    return f.v;
}

__device__ __forceinline__ v8f wmma_bf16(v16bf a, v16bf b, v8f c) {
    // D(f32 16x16) = A(bf16 16x32) * B(bf16 32x16) + C
    return __builtin_amdgcn_wmma_f32_16x16x32_bf16(false, a, false, b,
                                                   (short)0, c, false, false);
}

// Async DMA: copy 32B/lane (2 x b128) global -> LDS.  Per CDNA5 ISA the
// IOFFSET is added to BOTH the LDS address and the global address, so a single
// lds-addr VGPR + single 32-bit global offset VGPR covers the chunk.
__device__ __forceinline__ void async_copy32B(unsigned lds, unsigned long long base,
                                              unsigned voff) {
    asm volatile(
        "global_load_async_to_lds_b128 %0, %1, %2 offset:0\n\t"
        "global_load_async_to_lds_b128 %0, %1, %2 offset:16"
        :: "v"(lds), "v"(voff), "s"(base) : "memory");
}

__device__ __forceinline__ void async_copy16B(unsigned lds, unsigned long long base,
                                              unsigned voff) {
    asm volatile(
        "global_load_async_to_lds_b128 %0, %1, %2 offset:0"
        :: "v"(lds), "v"(voff), "s"(base) : "memory");
}

__device__ __forceinline__ void wait_async0() {
    asm volatile("s_wait_asynccnt 0x0" ::: "memory");
}

// Low 32 bits of a generic pointer to __shared__ == raw LDS byte offset
__device__ __forceinline__ unsigned lds_addr(const void* p) {
    return (unsigned)(unsigned long long)(uintptr_t)p;
}

// ---------------------------------------------------------------------------
// fp32 -> bf16 cast (4 elements / thread)
// ---------------------------------------------------------------------------
__global__ void cast_bf16_kernel(const float* __restrict__ src,
                                 __bf16* __restrict__ dst, int n) {
    int i = (blockIdx.x * blockDim.x + threadIdx.x) * 4;
    if (i < n) {
#pragma unroll
        for (int j = 0; j < 4; ++j) dst[i + j] = (__bf16)src[i + j];
    }
}

// ---------------------------------------------------------------------------
// W[K][N] fp32  ->  Wt[N][K] bf16   (32x32 LDS tile transpose)
// ---------------------------------------------------------------------------
__global__ void transpose_cast_kernel(const float* __restrict__ src,
                                      __bf16* __restrict__ dst, int K, int N) {
    __shared__ float tile[32][33];
    int n0 = blockIdx.x * 32, k0 = blockIdx.y * 32;
    int tx = threadIdx.x, ty = threadIdx.y;   // block (32, 8)
#pragma unroll
    for (int i = 0; i < 32; i += 8)
        tile[ty + i][tx] = src[(size_t)(k0 + ty + i) * N + n0 + tx];
    __syncthreads();
#pragma unroll
    for (int i = 0; i < 32; i += 8)
        dst[(size_t)(n0 + ty + i) * K + k0 + tx] = (__bf16)tile[tx][ty + i];
}

// ---------------------------------------------------------------------------
// GEMM: C[M][N] = A[M][K](bf16) * Bt[N][K](bf16)^T + bias, optional ReLU.
// Block tile 128x128, 256 threads = 8 waves (4x2), wave tile 32x64.
// K-step 32, double-buffered LDS filled by GLOBAL_LOAD_ASYNC_TO_LDS_B128.
// ---------------------------------------------------------------------------
__global__ void gemm_bf16_kernel(const __bf16* __restrict__ A,
                                 const __bf16* __restrict__ Bt,
                                 const float* __restrict__ bias,
                                 __bf16* __restrict__ Cbf,
                                 float* __restrict__ Cf,
                                 int M, int N, int Kd, int relu) {
    __shared__ __align__(16) __bf16 Alds[2][128][40];   // 32 K + 8 pad
    __shared__ __align__(16) __bf16 Blds[2][128][40];

    const int tid  = threadIdx.x;
    const int wave = tid >> 5, lane = tid & 31;
    const int wm = wave >> 1, wn = wave & 1;
    const int bm = blockIdx.y * 128, bn = blockIdx.x * 128;
    const int kb = (lane >> 4) * 8;
    const int lr = tid >> 1;          // 0..127 : tile row staged by this thread
    const int lc = (tid & 1) * 16;    // 0 or 16 (elements)

    const unsigned long long abase = (unsigned long long)(uintptr_t)A;
    const unsigned long long bbase = (unsigned long long)(uintptr_t)Bt;
    const unsigned avoff = (unsigned)(((size_t)(bm + lr) * Kd + lc) * 2);
    const unsigned bvoff = (unsigned)(((size_t)(bn + lr) * Kd + lc) * 2);
    const unsigned aldsv[2] = { lds_addr(&Alds[0][lr][lc]), lds_addr(&Alds[1][lr][lc]) };
    const unsigned bldsv[2] = { lds_addr(&Blds[0][lr][lc]), lds_addr(&Blds[1][lr][lc]) };

    v8f acc[2][4] = {};

    // prologue: DMA first K-tile into buffer 0
    async_copy32B(aldsv[0], abase, avoff);
    async_copy32B(bldsv[0], bbase, bvoff);

    const int nk = Kd >> 5;
    for (int it = 0; it < nk; ++it) {
        wait_async0();          // own DMA for buffer it&1 landed in LDS
        __syncthreads();        // everyone's landed; prev compute finished
        if (it + 1 < nk) {      // kick DMA for next tile into other buffer
            const unsigned koff = (unsigned)((it + 1) << 6);   // 32 elems * 2B
            async_copy32B(aldsv[(it + 1) & 1], abase, avoff + koff);
            async_copy32B(bldsv[(it + 1) & 1], bbase, bvoff + koff);
        }
        const __bf16 (*Ac)[40] = Alds[it & 1];
        const __bf16 (*Bc)[40] = Blds[it & 1];

        v16bf af[2], bf_[4];
#pragma unroll
        for (int i = 0; i < 2; ++i) {
            const __bf16* r = &Ac[wm * 32 + i * 16 + (lane & 15)][kb];
            af[i] = ldfrag(r, r + 16);
        }
#pragma unroll
        for (int j = 0; j < 4; ++j) {
            const __bf16* r = &Bc[wn * 64 + j * 16 + (lane & 15)][kb];
            bf_[j] = ldfrag(r, r + 16);
        }
#pragma unroll
        for (int i = 0; i < 2; ++i)
#pragma unroll
            for (int j = 0; j < 4; ++j)
                acc[i][j] = wmma_bf16(af[i], bf_[j], acc[i][j]);
    }

    // Epilogue. C-frag: vgpr r -> row = base + r + ((lane>>4)<<3); col = base + (lane&15)
#pragma unroll
    for (int i = 0; i < 2; ++i)
#pragma unroll
        for (int j = 0; j < 4; ++j) {
            const int col  = bn + wn * 64 + j * 16 + (lane & 15);
            const int rowb = bm + wm * 32 + i * 16 + ((lane >> 4) << 3);
            const float bv = bias[col];
#pragma unroll
            for (int r = 0; r < 8; ++r) {
                float v = acc[i][j][r] + bv;
                if (relu) v = fmaxf(v, 0.f);
                const size_t idx = (size_t)(rowb + r) * N + col;
                if (Cbf) Cbf[idx] = (__bf16)v;
                else     Cf[idx]  = v;
            }
        }
}

// ---------------------------------------------------------------------------
// Flash attention (non-causal), computing S^T = K*Q^T so softmax statistics
// are per-lane and the P-frag C->B conversion is element-wise per lane.
// Grid: (SEQ/128, N_HEADS, BATCH), block 256 (8 waves x 16 queries).
// K tile staged via async-to-LDS DMA; V tile transposed manually into LDS.
// ---------------------------------------------------------------------------
__global__ void flash_attn_kernel(const __bf16* __restrict__ Q,
                                  const __bf16* __restrict__ K,
                                  const __bf16* __restrict__ V,
                                  __bf16* __restrict__ O) {
    __shared__ __align__(16) __bf16 Klds[32][72];    // [key][d]
    __shared__ __align__(16) __bf16 Vtlds[64][40];   // [d][key]  (transposed)

    const int qblk = blockIdx.x, h = blockIdx.y, b = blockIdx.z;
    const int tid = threadIdx.x, wave = tid >> 5, lane = tid & 31;
    const int kb = (lane >> 4) * 8;
    const int ld = D_MODEL;

    const int qrow = qblk * 128 + wave * 16 + (lane & 15);
    const __bf16* qp = Q + ((size_t)(b * SEQ + qrow)) * ld + h * HEAD_DIM;
    // Q as B-operand of K*Q^T: lane = query col, K-dim = head dim
    const v16bf qf0 = ldfrag(qp +  0 + kb, qp +  0 + kb + 16);
    const v16bf qf1 = ldfrag(qp + 32 + kb, qp + 32 + kb + 16);

    v8f o0 = {}, o1 = {}, o2 = {}, o3 = {};
    float m = -1e30f, l = 0.f;
    const float scale = 0.125f;   // 1/sqrt(64)

    const int cr = tid >> 3;             // cooperative-load row   0..31
    const int cc = (tid & 7) * 8;        // cooperative-load col   0..56

    const unsigned long long kbase = (unsigned long long)(uintptr_t)
        (K + (size_t)(b * SEQ) * ld + h * HEAD_DIM);
    const unsigned kvoff0 = (unsigned)(((size_t)cr * ld + cc) * 2);
    const unsigned kldsv  = lds_addr(&Klds[cr][cc]);

    for (int kt = 0; kt < SEQ; kt += 32) {
        // DMA K tile [32][64] directly to LDS
        async_copy16B(kldsv, kbase, kvoff0 + (unsigned)(kt * ld * 2));
        // stage V^T tile [64][32] manually (needs element transpose)
        const size_t grow = (size_t)(b * SEQ + kt + cr) * ld + h * HEAD_DIM + cc;
        union { uint4 u; __bf16 e[8]; } vu;
        vu.u = *(const uint4*)(V + grow);
#pragma unroll
        for (int i = 0; i < 8; ++i) Vtlds[cc + i][cr] = vu.e[i];
        wait_async0();
        __syncthreads();

        const int r0 = lane & 15, r1 = 16 + r0;
        v16bf ka00 = ldfrag(&Klds[r0][kb],      &Klds[r0][kb + 16]);
        v16bf ka01 = ldfrag(&Klds[r0][32 + kb], &Klds[r0][32 + kb + 16]);
        v16bf ka10 = ldfrag(&Klds[r1][kb],      &Klds[r1][kb + 16]);
        v16bf ka11 = ldfrag(&Klds[r1][32 + kb], &Klds[r1][32 + kb + 16]);

        v8f c0 = {}, c1 = {};
        c0 = wmma_bf16(ka00, qf0, c0);  c0 = wmma_bf16(ka01, qf1, c0);
        c1 = wmma_bf16(ka10, qf0, c1);  c1 = wmma_bf16(ka11, qf1, c1);

        const int dr = lane & 15;
        v16bf va0 = ldfrag(&Vtlds[ 0 + dr][kb], &Vtlds[ 0 + dr][kb + 16]);
        v16bf va1 = ldfrag(&Vtlds[16 + dr][kb], &Vtlds[16 + dr][kb + 16]);
        v16bf va2 = ldfrag(&Vtlds[32 + dr][kb], &Vtlds[32 + dr][kb + 16]);
        v16bf va3 = ldfrag(&Vtlds[48 + dr][kb], &Vtlds[48 + dr][kb + 16]);

        // online softmax (per-lane query column; pair lanes L and L^16 share q)
        float loc = -1e30f;
#pragma unroll
        for (int r = 0; r < 8; ++r) {
            c0[r] *= scale; c1[r] *= scale;
            loc = fmaxf(loc, fmaxf(c0[r], c1[r]));
        }
        loc = fmaxf(loc, __shfl_xor(loc, 16, 32));
        const float mnew  = fmaxf(m, loc);
        const float alpha = __expf(m - mnew);
        float rs = 0.f;
        v16bf p;
#pragma unroll
        for (int r = 0; r < 8; ++r) {
            const float e0 = __expf(c0[r] - mnew);
            const float e1 = __expf(c1[r] - mnew);
            rs += e0 + e1;
            p[r]     = (__bf16)e0;   // keys kb-block of sub-tile 0
            p[8 + r] = (__bf16)e1;   // keys kb-block of sub-tile 1
        }
        rs += __shfl_xor(rs, 16, 32);
        l = l * alpha + rs;
        m = mnew;
#pragma unroll
        for (int r = 0; r < 8; ++r) {
            o0[r] *= alpha; o1[r] *= alpha; o2[r] *= alpha; o3[r] *= alpha;
        }
        // O^T += V^T * P^T
        o0 = wmma_bf16(va0, p, o0);
        o1 = wmma_bf16(va1, p, o1);
        o2 = wmma_bf16(va2, p, o2);
        o3 = wmma_bf16(va3, p, o3);
        __syncthreads();
    }

    const float inv = 1.f / l;
    __bf16* ob = O + ((size_t)(b * SEQ + qrow)) * ld + h * HEAD_DIM;
#pragma unroll
    for (int r = 0; r < 8; ++r) {
        const int dhi = r + ((lane >> 4) << 3);
        ob[ 0 + dhi] = (__bf16)(o0[r] * inv);
        ob[16 + dhi] = (__bf16)(o1[r] * inv);
        ob[32 + dhi] = (__bf16)(o2[r] * inv);
        ob[48 + dhi] = (__bf16)(o3[r] * inv);
    }
}

// ---------------------------------------------------------------------------
// y = LayerNorm(a + r) * g + b   (fp32 master, optional bf16 side output)
// ---------------------------------------------------------------------------
__global__ void ln_residual_kernel(const float* __restrict__ a,
                                   const float* __restrict__ rr,
                                   const float* __restrict__ g,
                                   const float* __restrict__ bb,
                                   float* __restrict__ of,
                                   __bf16* __restrict__ obf) {
    const int D = D_MODEL;
    const int row = blockIdx.x, tid = threadIdx.x;
    const float* pa = a  + (size_t)row * D;
    const float* pr = rr + (size_t)row * D;
    float v[4], s = 0.f, q = 0.f;
#pragma unroll
    for (int i = 0; i < 4; ++i) {
        const int c = tid + i * 256;
        const float x = pa[c] + pr[c];
        v[i] = x; s += x; q += x * x;
    }
    for (int o = 16; o > 0; o >>= 1) {
        s += __shfl_xor(s, o, 32);
        q += __shfl_xor(q, o, 32);
    }
    __shared__ float rs[8], rq[8];
    const int wave = tid >> 5, lane = tid & 31;
    if (lane == 0) { rs[wave] = s; rq[wave] = q; }
    __syncthreads();
    if (tid == 0) {
        float ts = 0.f, tq = 0.f;
        for (int i = 0; i < 8; ++i) { ts += rs[i]; tq += rq[i]; }
        rs[0] = ts; rq[0] = tq;
    }
    __syncthreads();
    const float mu  = rs[0] / D;
    const float var = rq[0] / D - mu * mu;
    const float inv = rsqrtf(var + 1e-5f);
#pragma unroll
    for (int i = 0; i < 4; ++i) {
        const int c = tid + i * 256;
        const float y = (v[i] - mu) * inv * g[c] + bb[c];
        of[(size_t)row * D + c] = y;
        if (obf) obf[(size_t)row * D + c] = (__bf16)y;
    }
}

// ---------------------------------------------------------------------------
extern "C" void kernel_launch(void* const* d_in, const int* in_sizes, int n_in,
                              void* d_out, int out_size, void* d_ws, size_t ws_size,
                              hipStream_t stream) {
    const float* x    = (const float*)d_in[0];
    const float* wq   = (const float*)d_in[1];
    const float* bq   = (const float*)d_in[2];
    const float* wk   = (const float*)d_in[3];
    const float* bk   = (const float*)d_in[4];
    const float* wv   = (const float*)d_in[5];
    const float* bv   = (const float*)d_in[6];
    const float* wo   = (const float*)d_in[7];
    const float* bo   = (const float*)d_in[8];
    const float* w1   = (const float*)d_in[9];
    const float* b1   = (const float*)d_in[10];
    const float* w2   = (const float*)d_in[11];
    const float* b2   = (const float*)d_in[12];
    const float* ln1g = (const float*)d_in[13];
    const float* ln1b = (const float*)d_in[14];
    const float* ln2g = (const float*)d_in[15];
    const float* ln2b = (const float*)d_in[16];

    char* ws = (char*)d_ws;
    const size_t MiB = 1024 * 1024;
    __bf16* WQT  = (__bf16*)(ws + 0 * MiB);    // 2 MiB each
    __bf16* WKT  = (__bf16*)(ws + 2 * MiB);
    __bf16* WVT  = (__bf16*)(ws + 4 * MiB);
    __bf16* WOT  = (__bf16*)(ws + 6 * MiB);
    __bf16* W1T  = (__bf16*)(ws + 8 * MiB);    // 8 MiB
    __bf16* W2T  = (__bf16*)(ws + 16 * MiB);   // 8 MiB
    __bf16* XBF  = (__bf16*)(ws + 24 * MiB);   // 8 MiB : x bf16, later attn-out bf16
    __bf16* QBF  = (__bf16*)(ws + 32 * MiB);   // 8 MiB
    __bf16* KBF  = (__bf16*)(ws + 40 * MiB);   // 8 MiB
    __bf16* VBF  = (__bf16*)(ws + 48 * MiB);   // 8 MiB
    float*  TMPF = (float* )(ws + 32 * MiB);   // 16 MiB : o-proj f32, later ffn2 f32
    float*  X1F  = (float* )(ws + 56 * MiB);   // 16 MiB
    __bf16* X1BF = (__bf16*)(ws + 72 * MiB);   // 8 MiB
    __bf16* HBF  = (__bf16*)(ws + 80 * MiB);   // 32 MiB
    float*  out  = (float*)d_out;

    const dim3 tb(32, 8);

    // 1) casts / weight transposes
    cast_bf16_kernel<<<(NTOK * D_MODEL) / 1024, 256, 0, stream>>>(x, XBF, NTOK * D_MODEL);
    transpose_cast_kernel<<<dim3(D_MODEL/32, D_MODEL/32), tb, 0, stream>>>(wq, WQT, D_MODEL, D_MODEL);
    transpose_cast_kernel<<<dim3(D_MODEL/32, D_MODEL/32), tb, 0, stream>>>(wk, WKT, D_MODEL, D_MODEL);
    transpose_cast_kernel<<<dim3(D_MODEL/32, D_MODEL/32), tb, 0, stream>>>(wv, WVT, D_MODEL, D_MODEL);
    transpose_cast_kernel<<<dim3(D_MODEL/32, D_MODEL/32), tb, 0, stream>>>(wo, WOT, D_MODEL, D_MODEL);
    transpose_cast_kernel<<<dim3(D_FF/32,    D_MODEL/32), tb, 0, stream>>>(w1, W1T, D_MODEL, D_FF);
    transpose_cast_kernel<<<dim3(D_MODEL/32, D_FF/32),    tb, 0, stream>>>(w2, W2T, D_FF, D_MODEL);

    // 2) QKV projections
    gemm_bf16_kernel<<<dim3(D_MODEL/128, NTOK/128), 256, 0, stream>>>(XBF, WQT, bq, QBF, nullptr, NTOK, D_MODEL, D_MODEL, 0);
    gemm_bf16_kernel<<<dim3(D_MODEL/128, NTOK/128), 256, 0, stream>>>(XBF, WKT, bk, KBF, nullptr, NTOK, D_MODEL, D_MODEL, 0);
    gemm_bf16_kernel<<<dim3(D_MODEL/128, NTOK/128), 256, 0, stream>>>(XBF, WVT, bv, VBF, nullptr, NTOK, D_MODEL, D_MODEL, 0);

    // 3) flash attention -> XBF (concat heads, row-major)
    flash_attn_kernel<<<dim3(SEQ/128, N_HEADS, BATCH), 256, 0, stream>>>(QBF, KBF, VBF, XBF);

    // 4) output projection (f32) + residual LN1
    gemm_bf16_kernel<<<dim3(D_MODEL/128, NTOK/128), 256, 0, stream>>>(XBF, WOT, bo, nullptr, TMPF, NTOK, D_MODEL, D_MODEL, 0);
    ln_residual_kernel<<<NTOK, 256, 0, stream>>>(x, TMPF, ln1g, ln1b, X1F, X1BF);

    // 5) FFN
    gemm_bf16_kernel<<<dim3(D_FF/128,    NTOK/128), 256, 0, stream>>>(X1BF, W1T, b1, HBF, nullptr, NTOK, D_FF, D_MODEL, 1);
    gemm_bf16_kernel<<<dim3(D_MODEL/128, NTOK/128), 256, 0, stream>>>(HBF,  W2T, b2, nullptr, TMPF, NTOK, D_MODEL, D_FF, 0);

    // 6) residual LN2 -> final output
    ln_residual_kernel<<<NTOK, 256, 0, stream>>>(X1F, TMPF, ln2g, ln2b, out, nullptr);
}
